// Attention4_69483980914987
// MI455X (gfx1250) — matrix operand order repaired and checked by reading, not compiled, not run on hardware
//
#include <hip/hip_runtime.h>
#include <hip/hip_bf16.h>

#define B_   64
#define S_   256
#define ED   300
#define EDP  320
#define H_   512
#define H3   1536
#define ATT_ 256

typedef __attribute__((ext_vector_type(16))) __bf16 v16bf;
typedef __attribute__((ext_vector_type(8)))  float  v8f;

// ---------------- WMMA fragment loaders (CDNA5 16x16x32 bf16 layouts) ----------------

// A: 16(M) x 32(K), row-major bf16 source, lda in elements.
// lanes 0-15: m=lane, v0..v7 hold K 0..15 (pairs)
// lanes 16-31: m=lane-16, v0..v7 hold K 16..31
__device__ __forceinline__ v16bf frag_a(const __bf16* __restrict__ A, int lda, int lane) {
    const int m   = lane & 15;
    const int khi = (lane & 16) ? 16 : 0;
    const __bf16* p = A + m * lda;
    v16bf a;
#pragma unroll
    for (int j = 0; j < 8; ++j) {
        int kk = khi + (j << 1);
        a[2 * j]     = p[kk];
        a[2 * j + 1] = p[kk + 1];
    }
    return a;
}

// B = W^T where W is row-major [N,K] (C = A @ W^T). B[k,n] = W[n,k].
// lanes 0-15: n=lane, VGPR j holds K = 2j,2j+1 ; lanes 16-31: K += 16
__device__ __forceinline__ v16bf frag_b_wt(const __bf16* __restrict__ W, int ldw, int lane) {
    const int n   = lane & 15;
    const int khi = (lane & 16) ? 16 : 0;
    const __bf16* p = W + n * ldw + khi;
    v16bf b;
#pragma unroll
    for (int j = 0; j < 8; ++j) {
        b[2 * j]     = p[2 * j];
        b[2 * j + 1] = p[2 * j + 1];
    }
    return b;
}

// B row-major [K,N] (non-transposed): B[k,n] = src[k*ldb + n]
__device__ __forceinline__ v16bf frag_b_nt(const __bf16* __restrict__ Bm, int ldb, int lane) {
    const int n   = lane & 15;
    const int khi = (lane & 16) ? 16 : 0;
    const __bf16* p = Bm + khi * ldb + n;
    v16bf b;
#pragma unroll
    for (int j = 0; j < 8; ++j) {
        b[2 * j]     = p[(2 * j) * ldb];
        b[2 * j + 1] = p[(2 * j + 1) * ldb];
    }
    return b;
}

// ---------------- generic bf16 WMMA GEMM: C = act(A @ op(B) + bias) ----------------
// Block: 256 thr = 8 waves; block covers 16 rows x 128 cols; grid = (N/128, M/16, batch)
template <bool TRANS_B, bool TANH>
__global__ __launch_bounds__(256) void gemm_bf16(
    const __bf16* __restrict__ A, int lda, long long sA,
    const __bf16* __restrict__ Bm, int ldb, long long sB,
    float* __restrict__ C, long long sC,
    __bf16* __restrict__ Cbf, int ldc,
    const float* __restrict__ bias,
    const float* __restrict__ bias_rg, int rows_per_group,
    int N, int K)
{
    __shared__ __bf16 As[16 * 1024];   // K <= 1024 for all uses
    const int z    = blockIdx.z;
    const int row0 = blockIdx.y * 16;
    const int tid  = threadIdx.x;

    const __bf16* Ab = A + (long long)z * sA + (long long)row0 * lda;
    for (int i = tid; i < 16 * K; i += 256)
        As[i] = Ab[(i / K) * lda + (i % K)];
    __syncthreads();

    const int wave = tid >> 5, lane = tid & 31;
    const int n0   = blockIdx.x * 128 + wave * 16;
    const __bf16* Bz = Bm + (long long)z * sB;

    v8f acc = {};
    for (int k0 = 0; k0 < K; k0 += 32) {
        v16bf a = frag_a(As + k0, K, lane);
        v16bf b;
        if (TRANS_B) b = frag_b_wt(Bz + (long long)n0 * ldb + k0, ldb, lane);
        else         b = frag_b_nt(Bz + (long long)k0 * ldb + n0, ldb, lane);
        acc = __builtin_amdgcn_wmma_f32_16x16x32_bf16(false, a, false, b,
                                                      (short)0, acc, false, false);
    }

    const int col = lane & 15;
    const int rlo = (lane & 16) ? 8 : 0;
#pragma unroll
    for (int i = 0; i < 8; ++i) {
        int r = row0 + rlo + i;
        int n = n0 + col;
        float v = acc[i];
        if (bias)    v += bias[n];
        if (bias_rg) v += bias_rg[(r / rows_per_group) * N + n];
        if (TANH)    v = tanhf(v);
        long long idx = (long long)z * sC + (long long)r * ldc + n;
        if (C)   C[idx]   = v;
        if (Cbf) Cbf[idx] = (__bf16)v;
    }
}

// ---------------- software grid barrier (persistent GRU kernel) ----------------
__device__ __forceinline__ void grid_barrier(unsigned* bar, unsigned target) {
    __syncthreads();
    if (threadIdx.x == 0) {
        __threadfence();
        atomicAdd(bar, 1u);
        while (atomicAdd(bar, 0u) < target) __builtin_amdgcn_s_sleep(2);
        __threadfence();
    }
    __syncthreads();
}

__device__ __forceinline__ float sigmoidf_(float x) { return 1.0f / (1.0f + expf(-x)); }

// ---------------- persistent bidirectional GRU ----------------
// grid (16, 2): dir = blockIdx.y; m_tile = blockIdx.x & 3 (16 batch rows);
// ngrp = blockIdx.x >> 2 ; wave handles n-tile jt = ngrp*8+wave (cols j0..j0+15 of H)
// Each wave accumulates r/z/n gate tiles (3 x 16x16, K=512) -> gate math in regs.
__global__ __launch_bounds__(256) void gru_kernel(
    const float* __restrict__ xw_f, const float* __restrict__ xw_b,
    const __bf16* __restrict__ Whh_f, const __bf16* __restrict__ Whh_b,
    const float* __restrict__ bhh_f, const float* __restrict__ bhh_b,
    float* __restrict__ hstate, __bf16* __restrict__ hstate_bf,
    float* __restrict__ hcat, __bf16* __restrict__ hcat_bf,
    unsigned* __restrict__ bar)
{
    __shared__ __bf16 Hs[16 * H_];
    const int dir    = blockIdx.y;
    const int m_tile = blockIdx.x & 3;
    const int ngrp   = blockIdx.x >> 2;
    const int tid    = threadIdx.x;
    const int wave   = tid >> 5, lane = tid & 31;
    const int j0     = (ngrp * 8 + wave) * 16;

    const float*  xw  = dir ? xw_b  : xw_f;
    const __bf16* Whh = dir ? Whh_b : Whh_f;
    const float*  bhh = dir ? bhh_b : bhh_f;
    float*  hs  = hstate    + dir * (B_ * H_);
    __bf16* hsb = hstate_bf + dir * (B_ * H_);
    const unsigned NBLK = gridDim.x * gridDim.y;   // 32

    // init h = 0
    if (ngrp == 0) {
        for (int i = tid; i < 16 * H_; i += 256) {
            int r = m_tile * 16 + (i >> 9), c = i & 511;
            hs[r * H_ + c]  = 0.0f;
            hsb[r * H_ + c] = (__bf16)0.0f;
        }
    }
    grid_barrier(bar, NBLK);

    for (int step = 0; step < S_; ++step) {
        const int t = dir ? (S_ - 1 - step) : step;

        // stage this m-tile's h (bf16) into LDS
        for (int i = tid; i < 16 * H_; i += 256)
            Hs[i] = hsb[(m_tile * 16 + (i >> 9)) * H_ + (i & 511)];
        __syncthreads();

        v8f ar = {}, az = {}, an = {};
#pragma unroll 4
        for (int kc = 0; kc < H_; kc += 32) {
            v16bf a  = frag_a(Hs + kc, H_, lane);
            v16bf br = frag_b_wt(Whh + (long long)(j0)            * H_ + kc, H_, lane);
            v16bf bz = frag_b_wt(Whh + (long long)(H_ + j0)       * H_ + kc, H_, lane);
            v16bf bn = frag_b_wt(Whh + (long long)(2 * H_ + j0)   * H_ + kc, H_, lane);
            ar = __builtin_amdgcn_wmma_f32_16x16x32_bf16(false, a, false, br, (short)0, ar, false, false);
            az = __builtin_amdgcn_wmma_f32_16x16x32_bf16(false, a, false, bz, (short)0, az, false, false);
            an = __builtin_amdgcn_wmma_f32_16x16x32_bf16(false, a, false, bn, (short)0, an, false, false);
        }

        const int col = lane & 15;
        const int rlo = (lane & 16) ? 8 : 0;
        const int j   = j0 + col;
        const float bh_r = bhh[j], bh_z = bhh[H_ + j], bh_n = bhh[2 * H_ + j];
#pragma unroll
        for (int i = 0; i < 8; ++i) {
            int rb = m_tile * 16 + rlo + i;                       // batch row
            long long xrow = ((long long)rb * S_ + t) * H3;
            float xr = xw[xrow + j], xz = xw[xrow + H_ + j], xn = xw[xrow + 2 * H_ + j];
            float r  = sigmoidf_(xr + ar[i] + bh_r);
            float zg = sigmoidf_(xz + az[i] + bh_z);
            float nn = tanhf(xn + r * (an[i] + bh_n));
            float hold = hs[rb * H_ + j];
            float hnew = (1.0f - zg) * nn + zg * hold;
            hs[rb * H_ + j]  = hnew;
            hsb[rb * H_ + j] = (__bf16)hnew;
            long long orow = ((long long)rb * S_ + t) * (2 * H_) + dir * H_ + j;
            hcat[orow]    = hnew;
            hcat_bf[orow] = (__bf16)hnew;
        }
        __syncthreads();
        grid_barrier(bar, NBLK * (unsigned)(step + 2));
    }
}

// ---------------- embedding gather + max-norm renorm -> bf16 padded [BS, 320] ----------------
__global__ __launch_bounds__(256) void embed_kernel(const int* __restrict__ x,
    const float* __restrict__ emb, __bf16* __restrict__ e_out)
{
    const int tok  = blockIdx.x * 8 + (threadIdx.x >> 5);
    const int lane = threadIdx.x & 31;
    const float* row = emb + (long long)x[tok] * ED;
    float v[10]; float ss = 0.f;
#pragma unroll
    for (int i = 0; i < 10; ++i) {
        int d = lane + 32 * i;
        float t = (d < ED) ? row[d] : 0.f;
        v[i] = t; ss += t * t;
    }
#pragma unroll
    for (int o = 16; o > 0; o >>= 1) ss += __shfl_xor(ss, o, 32);
    float sc = fminf(1.0f, 5.0f / (sqrtf(ss) + 1e-7f));
    __bf16* dst = e_out + (long long)tok * EDP;
#pragma unroll
    for (int i = 0; i < 10; ++i) {
        int d = lane + 32 * i;
        dst[d] = (__bf16)((d < ED) ? v[i] * sc : 0.f);   // pads 300..319 with 0
    }
}

// ---------------- masked mean target over [ts, te] ----------------
__global__ void target_kernel(const float* __restrict__ hcat, const int* __restrict__ ts,
    const int* __restrict__ te, float* __restrict__ tgt, __bf16* __restrict__ tgt_bf)
{
    const int b = blockIdx.x;
    const int t0 = ts[b], t1 = te[b];
    const float inv = 1.0f / (float)(t1 - t0 + 1);
    for (int c = threadIdx.x; c < 2 * H_; c += 256) {
        float s = 0.f;
        for (int t = t0; t <= t1; ++t) s += hcat[((long long)b * S_ + t) * (2 * H_) + c];
        s *= inv;
        tgt[b * (2 * H_) + c]    = s;
        tgt_bf[b * (2 * H_) + c] = (__bf16)s;
    }
}

// ---------------- softmax over s (row length 256), one wave per row ----------------
__global__ __launch_bounds__(256) void softmax_kernel(const float* __restrict__ beta,
    __bf16* __restrict__ alfa)
{
    const int row  = blockIdx.x * 8 + (threadIdx.x >> 5);
    const int lane = threadIdx.x & 31;
    const float* p = beta + (long long)row * S_;
    float v[8]; float mx = -3.4e38f;
#pragma unroll
    for (int i = 0; i < 8; ++i) { v[i] = p[lane + 32 * i]; mx = fmaxf(mx, v[i]); }
    for (int o = 16; o > 0; o >>= 1) mx = fmaxf(mx, __shfl_xor(mx, o, 32));
    float sum = 0.f;
#pragma unroll
    for (int i = 0; i < 8; ++i) { v[i] = expf(v[i] - mx); sum += v[i]; }
    for (int o = 16; o > 0; o >>= 1) sum += __shfl_xor(sum, o, 32);
    float inv = 1.0f / sum;
    __bf16* q = alfa + (long long)row * S_;
#pragma unroll
    for (int i = 0; i < 8; ++i) q[lane + 32 * i] = (__bf16)(v[i] * inv);
}

// ---------------- final tiny GEMM: out = result @ W2^T + b2 (N=3) ----------------
__global__ void final_kernel(const float* __restrict__ res, const float* __restrict__ W2,
    const float* __restrict__ b2, float* __restrict__ out)
{
    int idx = blockIdx.x * blockDim.x + threadIdx.x;   // (b*256 + k)
    if (idx >= B_ * ATT_) return;
    const float* r = res + (long long)idx * (2 * H_);
    float a0 = b2[0], a1 = b2[1], a2 = b2[2];
    for (int h = 0; h < 2 * H_; ++h) {
        float v = r[h];
        a0 += v * W2[h];
        a1 += v * W2[2 * H_ + h];
        a2 += v * W2[2 * (2 * H_) + h];
    }
    out[idx * 3 + 0] = a0; out[idx * 3 + 1] = a1; out[idx * 3 + 2] = a2;
}

// ---------------- f32 -> bf16 weight conversion with column slice/pad ----------------
__global__ void cvt_kernel(const float* __restrict__ src, int sstride, int soff,
    __bf16* __restrict__ dst, int dcols, int ccols, int total)
{
    int i = blockIdx.x * blockDim.x + threadIdx.x;
    if (i >= total) return;
    int r = i / dcols, c = i % dcols;
    dst[i] = (__bf16)((c < ccols) ? src[(long long)r * sstride + soff + c] : 0.0f);
}

__global__ void zero_kernel(unsigned* __restrict__ p, int n) {
    int i = blockIdx.x * blockDim.x + threadIdx.x;
    if (i < n) p[i] = 0u;
}

// ---------------- launcher ----------------
extern "C" void kernel_launch(void* const* d_in, const int* in_sizes, int n_in,
                              void* d_out, int out_size, void* d_ws, size_t ws_size,
                              hipStream_t stream)
{
    (void)in_sizes; (void)n_in; (void)out_size; (void)ws_size;
    const int*   x     = (const int*)d_in[0];
    const int*   tst   = (const int*)d_in[1];
    const int*   ten   = (const int*)d_in[2];
    const float* emb   = (const float*)d_in[3];
    const float* Wih_f = (const float*)d_in[4];
    const float* Whh_f = (const float*)d_in[5];
    const float* bih_f = (const float*)d_in[6];
    const float* bhh_f = (const float*)d_in[7];
    const float* Wih_b = (const float*)d_in[8];
    const float* Whh_b = (const float*)d_in[9];
    const float* bih_b = (const float*)d_in[10];
    const float* bhh_b = (const float*)d_in[11];
    const float* W1    = (const float*)d_in[12];
    const float* b1    = (const float*)d_in[13];
    const float* u     = (const float*)d_in[14];
    const float* W2    = (const float*)d_in[15];
    const float* b2    = (const float*)d_in[16];
    float* out = (float*)d_out;

    char* ws = (char*)d_ws;
    size_t off = 0;
    auto alloc = [&](size_t bytes) -> void* {
        void* p = ws + off;
        off += (bytes + 255) & ~(size_t)255;
        return p;
    };
    const long long BS = (long long)B_ * S_;
    __bf16* wihf  = (__bf16*)alloc((size_t)H3 * EDP * 2);
    __bf16* wihb  = (__bf16*)alloc((size_t)H3 * EDP * 2);
    __bf16* whhf  = (__bf16*)alloc((size_t)H3 * H_ * 2);
    __bf16* whhb  = (__bf16*)alloc((size_t)H3 * H_ * 2);
    __bf16* w1a   = (__bf16*)alloc((size_t)ATT_ * 1024 * 2);
    __bf16* w1b   = (__bf16*)alloc((size_t)ATT_ * 1024 * 2);
    __bf16* u_bf  = (__bf16*)alloc((size_t)ATT_ * ATT_ * 2);
    __bf16* e_bf  = (__bf16*)alloc((size_t)BS * EDP * 2);
    float*  xwf   = (float*) alloc((size_t)BS * H3 * 4);
    float*  xwb   = (float*) alloc((size_t)BS * H3 * 4);
    float*  hst   = (float*) alloc((size_t)2 * B_ * H_ * 4);
    __bf16* hstb  = (__bf16*)alloc((size_t)2 * B_ * H_ * 2);
    float*  hcat  = (float*) alloc((size_t)BS * 2 * H_ * 4);
    __bf16* hcatb = (__bf16*)alloc((size_t)BS * 2 * H_ * 2);
    float*  tgt   = (float*) alloc((size_t)B_ * 2 * H_ * 4);
    __bf16* tgtb  = (__bf16*)alloc((size_t)B_ * 2 * H_ * 2);
    float*  tcon  = (float*) alloc((size_t)B_ * ATT_ * 4);
    __bf16* o_bf  = (__bf16*)alloc((size_t)BS * ATT_ * 2);
    float*  beta  = (float*) alloc((size_t)B_ * ATT_ * S_ * 4);
    __bf16* alfa  = (__bf16*)alloc((size_t)B_ * ATT_ * S_ * 2);
    float*  resu  = (float*) alloc((size_t)B_ * ATT_ * 2 * H_ * 4);
    unsigned* bar = (unsigned*)alloc(256);

    dim3 blk(256);
    int n;
    zero_kernel<<<1, 64, 0, stream>>>(bar, 64);

    // weight conversions (f32 -> bf16, pad K where needed)
    n = H3 * EDP;
    cvt_kernel<<<(n + 255) / 256, blk, 0, stream>>>(Wih_f, ED, 0, wihf, EDP, ED, n);
    cvt_kernel<<<(n + 255) / 256, blk, 0, stream>>>(Wih_b, ED, 0, wihb, EDP, ED, n);
    n = H3 * H_;
    cvt_kernel<<<(n + 255) / 256, blk, 0, stream>>>(Whh_f, H_, 0, whhf, H_, H_, n);
    cvt_kernel<<<(n + 255) / 256, blk, 0, stream>>>(Whh_b, H_, 0, whhb, H_, H_, n);
    n = ATT_ * 1024;
    cvt_kernel<<<(n + 255) / 256, blk, 0, stream>>>(W1, 2048, 0,    w1a, 1024, 1024, n);
    cvt_kernel<<<(n + 255) / 256, blk, 0, stream>>>(W1, 2048, 1024, w1b, 1024, 1024, n);
    n = ATT_ * ATT_;
    cvt_kernel<<<(n + 255) / 256, blk, 0, stream>>>(u, ATT_, 0, u_bf, ATT_, ATT_, n);

    // embedding gather + renorm
    embed_kernel<<<(B_ * S_) / 8, blk, 0, stream>>>(x, emb, e_bf);

    // input projections xw = e @ Wih^T + bih (both directions), M=16384 N=1536 K=320
    gemm_bf16<true, false><<<dim3(H3 / 128, (B_ * S_) / 16, 1), blk, 0, stream>>>(
        e_bf, EDP, 0, wihf, EDP, 0, xwf, 0, nullptr, H3, bih_f, nullptr, 1, H3, EDP);
    gemm_bf16<true, false><<<dim3(H3 / 128, (B_ * S_) / 16, 1), blk, 0, stream>>>(
        e_bf, EDP, 0, wihb, EDP, 0, xwb, 0, nullptr, H3, bih_b, nullptr, 1, H3, EDP);

    // bidirectional GRU (persistent, grid-barrier per time step)
    gru_kernel<<<dim3(16, 2), blk, 0, stream>>>(xwf, xwb, whhf, whhb, bhh_f, bhh_b,
                                                hst, hstb, hcat, hcatb, bar);

    // masked mean target
    target_kernel<<<B_, blk, 0, stream>>>(hcat, tst, ten, tgt, tgtb);

    // t_contrib = target @ W1[:,1024:]^T + b1    (M=64 N=256 K=1024)
    gemm_bf16<true, false><<<dim3(ATT_ / 128, B_ / 16, 1), blk, 0, stream>>>(
        tgtb, 2 * H_, 0, w1b, 1024, 0, tcon, 0, nullptr, ATT_, b1, nullptr, 1, ATT_, 1024);

    // o = tanh(h @ W1[:,:1024]^T + t_contrib[b])  (M=16384 N=256 K=1024) -> bf16
    gemm_bf16<true, true><<<dim3(ATT_ / 128, (B_ * S_) / 16, 1), blk, 0, stream>>>(
        hcatb, 2 * H_, 0, w1a, 1024, 0, nullptr, 0, o_bf, ATT_, nullptr, tcon, S_, ATT_, 1024);

    // beta[b] = u @ o[b]^T   (batched: M=256 N=256 K=256)
    gemm_bf16<true, false><<<dim3(S_ / 128, ATT_ / 16, B_), blk, 0, stream>>>(
        u_bf, ATT_, 0, o_bf, ATT_, (long long)ATT_ * S_, beta, (long long)ATT_ * S_,
        nullptr, S_, nullptr, nullptr, 1, S_, ATT_);

    // softmax over s -> alfa (bf16)
    softmax_kernel<<<(B_ * ATT_) / 8, blk, 0, stream>>>(beta, alfa);

    // result[b] = alfa[b] @ h[b]   (batched: M=256 N=1024 K=256, B non-transposed)
    gemm_bf16<false, false><<<dim3((2 * H_) / 128, ATT_ / 16, B_), blk, 0, stream>>>(
        alfa, S_, (long long)ATT_ * S_, hcatb, 2 * H_, (long long)S_ * 2 * H_,
        resu, (long long)ATT_ * 2 * H_, nullptr, 2 * H_, nullptr, nullptr, 1, 2 * H_, S_);

    // out = result @ W2^T + b2
    final_kernel<<<(B_ * ATT_ + 255) / 256, blk, 0, stream>>>(resu, W2, b2, out);
}
